// MuVAN_4483945857463
// MI455X (gfx1250) — matrix operand order repaired and checked by reading, not compiled, not run on hardware
//
#include <hip/hip_runtime.h>
#include <hip/hip_bf16.h>
#include <stdint.h>

// ---------------------------------------------------------------------------
// Types / helpers for CDNA5 WMMA (wave32, 16x16x32 bf16 -> f32)
// ---------------------------------------------------------------------------
typedef __attribute__((ext_vector_type(16))) __bf16 v16bf;
typedef __attribute__((ext_vector_type(8)))  float  v8f;

struct alignas(16) U128 { unsigned int a, b, c, d; };

union BF16x16 {
  v16bf v;
  unsigned short u[16];
  U128 q[2];
};

__device__ __forceinline__ unsigned short f2bf(float f) {
  return (unsigned short)(__float_as_uint(f) >> 16);   // truncating f32->bf16
}
__device__ __forceinline__ float sigf(float x) { return 1.0f / (1.0f + __expf(-x)); }
// branch-free tanh (avoids ocml exec-mask branches in the serial cell update)
__device__ __forceinline__ float tanh_fast(float x) {
  float e = __expf(2.0f * x);
  return 1.0f - 2.0f / (e + 1.0f);
}

__device__ __forceinline__ v8f wmma_bf16(v16bf a, v16bf b, v8f c) {
  // 8 args: (neg_a, A, neg_b, B, c_mod, C, reuse_a, reuse_b)
  return __builtin_amdgcn_wmma_f32_16x16x32_bf16(false, a, false, b, (short)0, c,
                                                 false, false);
}

// A fragment (16x32 bf16). lanes 0-15 row M=lane hold K [kbase..+15];
// lanes 16-31 same rows, K [kbase+16..+31].
__device__ __forceinline__ v16bf load_a_lds(const unsigned short* h, int kbase, int lane) {
  int m  = lane & 15;
  int k0 = kbase + ((lane & 16) ? 16 : 0);
  const unsigned short* p = h + m * 128 + k0;
  BF16x16 r;
  r.q[0] = *(const U128*)(p);        // ds_load_b128
  r.q[1] = *(const U128*)(p + 8);
  return r.v;
}
__device__ __forceinline__ v16bf load_a_g(const unsigned short* x, int ldk, int kbase, int lane) {
  int m  = lane & 15;
  int k0 = kbase + ((lane & 16) ? 16 : 0);
  const unsigned short* p = x + (size_t)m * ldk + k0;
  BF16x16 r;
  r.q[0] = *(const U128*)(p);        // global_load_b128
  r.q[1] = *(const U128*)(p + 8);
  return r.v;
}
// B fragment (32x16 bf16) from W[n][k] (n-major). Lane holds column n = nbase+(l&15),
// 16 contiguous K values: lanes 0-15 K=[kbase..+15], lanes 16-31 K=[kbase+16..+31].
__device__ __forceinline__ v16bf load_b_bf16g(const unsigned short* w, int nbase, int ldk,
                                              int kbase, int lane) {
  int n  = nbase + (lane & 15);
  int k0 = kbase + ((lane & 16) ? 16 : 0);
  const unsigned short* p = w + (size_t)n * ldk + k0;
  BF16x16 r;
  r.q[0] = *(const U128*)(p);
  r.q[1] = *(const U128*)(p + 8);
  return r.v;
}
__device__ __forceinline__ v16bf load_b_f32(const float* w, int nbase, int ldk,
                                            int kbase, int lane) {
  int n  = nbase + (lane & 15);
  int k0 = kbase + ((lane & 16) ? 16 : 0);
  const float* p = w + (size_t)n * ldk + k0;
  BF16x16 r;
#pragma unroll
  for (int i = 0; i < 16; ++i) r.u[i] = f2bf(p[i]);
  return r.v;
}

// ---------------------------------------------------------------------------
// Model constants
// ---------------------------------------------------------------------------
#define S_LEN 128      // scan length (batch axis of reference)
#define NB    2048     // LSTM batch = V*T
#define HID   128
#define GATES 512
#define DDIM  256
#define VV    16
#define TT    128

__global__ void f32_to_bf16_k(const float* __restrict__ in, unsigned short* __restrict__ out, int n) {
  int i = blockIdx.x * blockDim.x + threadIdx.x;
  if (i < n) out[i] = f2bf(in[i]);
}

// ---------------------------------------------------------------------------
// LSTM layer 1 (input dim 1). One block = 16 batch rows, 8 waves.
// Gate-column remap: wave w owns hidden columns j in [w*16, w*16+16) of ALL
// FOUR gates (n-tile g -> cols g*128 + w*16). After the WMMAs, acc[0..3][r]
// hold i/f/g/o for the SAME (row, col) -> the cell update is pure registers:
// no gates-LDS round trip, one barrier/step (double-buffered h in LDS).
// Only the bf16 hidden state is written out (f32 copy is never consumed).
// blockIdx.x: row tile (0..127), blockIdx.y: direction (0 fwd, 1 bwd)
// ---------------------------------------------------------------------------
__global__ __launch_bounds__(256)
void lstm_l1_k(const float* __restrict__ x,           // (S, T, V)
               const float* __restrict__ whh_f, const float* __restrict__ whh_b,
               const float* __restrict__ wih_f, const float* __restrict__ wih_b,
               const float* __restrict__ bih_f, const float* __restrict__ bhh_f,
               const float* __restrict__ bih_b, const float* __restrict__ bhh_b,
               unsigned short* __restrict__ out_bf16) // (S, NB, 256)
{
  const int dir = blockIdx.y;
  const int rowbase = blockIdx.x * 16;
  const float* whh = dir ? whh_b : whh_f;
  const float* wih = dir ? wih_b : wih_f;
  const float* bih = dir ? bih_b : bih_f;
  const float* bhh = dir ? bhh_b : bhh_f;
  const int dcol = dir ? HID : 0;

  __shared__ alignas(16) unsigned short h_lds[2][16 * HID];   // double buffer

  const int tid  = threadIdx.x;
  const int lane = tid & 31;
  const int wv   = tid >> 5;
  const int jcol = wv * 16 + (lane & 15);   // this lane's hidden column
  const int mo   = (lane & 16) ? 8 : 0;     // accumulator row offset

  for (int i = tid; i < 16 * HID; i += 256) h_lds[0][i] = 0;

  // Whh B fragments: 4 gates x 4 k-steps, pinned in VGPRs for the whole scan
  v16bf bw[4][4];
#pragma unroll
  for (int g = 0; g < 4; ++g)
#pragma unroll
    for (int k = 0; k < 4; ++k)
      bw[g][k] = load_b_f32(whh, g * HID + wv * 16, HID, k * 32, lane);

  // per-lane biases and input weights for the 4 gates of column jcol
  float bsum[4], wx[4];
#pragma unroll
  for (int g = 0; g < 4; ++g) {
    int n = g * HID + jcol;
    bsum[g] = bih[n] + bhh[n];
    wx[g]   = wih[n];                       // Wih is (512, 1)
  }

  float creg[8];
#pragma unroll
  for (int r = 0; r < 8; ++r) creg[r] = 0.0f;

  const int v_idx = rowbase >> 7;           // view (constant per block)
  const int t0    = rowbase & 127;          // base time index

  __syncthreads();

  int cur = 0;
  for (int step = 0; step < S_LEN; ++step) {
    const int s = dir ? (S_LEN - 1 - step) : step;

    v8f zero = {};
    v8f acc[4];
#pragma unroll
    for (int g = 0; g < 4; ++g) acc[g] = zero;
#pragma unroll
    for (int k = 0; k < 4; ++k) {
      v16bf a = load_a_lds(h_lds[cur], k * 32, lane);
#pragma unroll
      for (int g = 0; g < 4; ++g)
        acc[g] = wmma_bf16(a, bw[g][k], acc[g]);
    }

    const float* xs = x + (size_t)s * 2048 + t0 * VV + v_idx;
    const int nxt = cur ^ 1;
#pragma unroll
    for (int r = 0; r < 8; ++r) {
      const int m = r + mo;
      float xv = xs[m * VV];
      float gi = acc[0][r] + bsum[0] + xv * wx[0];
      float gf = acc[1][r] + bsum[1] + xv * wx[1];
      float gg = acc[2][r] + bsum[2] + xv * wx[2];
      float go = acc[3][r] + bsum[3] + xv * wx[3];
      float cn = sigf(gf) * creg[r] + sigf(gi) * tanh_fast(gg);
      creg[r] = cn;
      float hv = sigf(go) * tanh_fast(cn);
      unsigned short hb = f2bf(hv);
      h_lds[nxt][m * HID + jcol] = hb;
      out_bf16[((size_t)s * NB + rowbase + m) * DDIM + dcol + jcol] = hb;
    }
    __syncthreads();
    cur = nxt;
  }
}

// ---------------------------------------------------------------------------
// LSTM layer 2 (input dim 256 = layer-1 bf16 output). Same register-resident
// cell update; the x @ Wih^T contribution streams bf16 Wih fragments from
// global (L2-resident) each step; next step's x tile is prefetched
// (global_prefetch_b8) to overlap with the WMMA burst.
// ---------------------------------------------------------------------------
__global__ __launch_bounds__(256)
void lstm_l2_k(const unsigned short* __restrict__ x2bf,   // (S, NB, 256) bf16
               const float* __restrict__ whh_f, const float* __restrict__ whh_b,
               const unsigned short* __restrict__ wihbf_f,
               const unsigned short* __restrict__ wihbf_b, // (512,256) bf16
               const float* __restrict__ bih_f, const float* __restrict__ bhh_f,
               const float* __restrict__ bih_b, const float* __restrict__ bhh_b,
               float* __restrict__ out_f32)                // (S, NB, 256)
{
  const int dir = blockIdx.y;
  const int rowbase = blockIdx.x * 16;
  const float* whh = dir ? whh_b : whh_f;
  const unsigned short* wih = dir ? wihbf_b : wihbf_f;
  const float* bih = dir ? bih_b : bih_f;
  const float* bhh = dir ? bhh_b : bhh_f;
  const int dcol = dir ? HID : 0;

  __shared__ alignas(16) unsigned short h_lds[2][16 * HID];

  const int tid  = threadIdx.x;
  const int lane = tid & 31;
  const int wv   = tid >> 5;
  const int jcol = wv * 16 + (lane & 15);
  const int mo   = (lane & 16) ? 8 : 0;

  for (int i = tid; i < 16 * HID; i += 256) h_lds[0][i] = 0;

  v16bf bw[4][4];
#pragma unroll
  for (int g = 0; g < 4; ++g)
#pragma unroll
    for (int k = 0; k < 4; ++k)
      bw[g][k] = load_b_f32(whh, g * HID + wv * 16, HID, k * 32, lane);

  float bsum[4];
#pragma unroll
  for (int g = 0; g < 4; ++g) {
    int n = g * HID + jcol;
    bsum[g] = bih[n] + bhh[n];
  }

  float creg[8];
#pragma unroll
  for (int r = 0; r < 8; ++r) creg[r] = 0.0f;

  __syncthreads();

  int cur = 0;
  for (int step = 0; step < S_LEN; ++step) {
    const int s = dir ? (S_LEN - 1 - step) : step;

    // prefetch next step's input tile (8KB) while this step's WMMAs run
    if (step + 1 < S_LEN) {
      const int sn = dir ? (S_LEN - 2 - step) : (step + 1);
      const unsigned short* xn = x2bf + ((size_t)sn * NB + rowbase) * DDIM;
      if (tid < 128) __builtin_prefetch(xn + tid * 32, 0, 1);
    }

    v8f zero = {};
    v8f acc[4];
#pragma unroll
    for (int g = 0; g < 4; ++g) acc[g] = zero;

    // h @ Whh^T  (K = 128)
#pragma unroll
    for (int k = 0; k < 4; ++k) {
      v16bf a = load_a_lds(h_lds[cur], k * 32, lane);
#pragma unroll
      for (int g = 0; g < 4; ++g)
        acc[g] = wmma_bf16(a, bw[g][k], acc[g]);
    }
    // x_t @ Wih^T  (K = 256), streamed B fragments
    const unsigned short* xrow = x2bf + ((size_t)s * NB + rowbase) * DDIM;
#pragma unroll
    for (int k2 = 0; k2 < 8; ++k2) {
      v16bf a2 = load_a_g(xrow, DDIM, k2 * 32, lane);
#pragma unroll
      for (int g = 0; g < 4; ++g) {
        v16bf b2 = load_b_bf16g(wih, g * HID + wv * 16, DDIM, k2 * 32, lane);
        acc[g] = wmma_bf16(a2, b2, acc[g]);
      }
    }

    const int nxt = cur ^ 1;
#pragma unroll
    for (int r = 0; r < 8; ++r) {
      const int m = r + mo;
      float gi = acc[0][r] + bsum[0];
      float gf = acc[1][r] + bsum[1];
      float gg = acc[2][r] + bsum[2];
      float go = acc[3][r] + bsum[3];
      float cn = sigf(gf) * creg[r] + sigf(gi) * tanh_fast(gg);
      creg[r] = cn;
      float hv = sigf(go) * tanh_fast(cn);
      h_lds[nxt][m * HID + jcol] = f2bf(hv);
      out_f32[((size_t)s * NB + rowbase + m) * DDIM + dcol + jcol] = hv;
    }
    __syncthreads();
    cur = nxt;
  }
}

// ---------------------------------------------------------------------------
// Tail kernels (attention + focus + conv + affine): VALU, negligible FLOPs.
// hidden[b,v,t,d] == h2[((b*2048 + v*128 + t)*256 + d]
// ---------------------------------------------------------------------------
__global__ void sum_last_k(const float* __restrict__ h2, float* __restrict__ out) {
  int b = blockIdx.x, d = threadIdx.x;           // (128, 256)
  float s = 0.f;
  for (int v = 0; v < VV; ++v) s += h2[((size_t)b * NB + v * TT + (TT - 1)) * DDIM + d];
  out[b * DDIM + d] = s;
}

__global__ void sum_i_k(const float* __restrict__ h2, float* __restrict__ out) {
  size_t idx = (size_t)blockIdx.x * blockDim.x + threadIdx.x;  // (B,127,256)
  if (idx >= (size_t)128 * 127 * 256) return;
  int d = idx & 255;
  size_t r = idx >> 8;
  int t = (int)(r % 127), b = (int)(r / 127);
  float s = 0.f;
  for (int v = 0; v < VV; ++v) s += h2[((size_t)b * NB + v * TT + t) * DDIM + d];
  out[idx] = s;
}

// sliding dot-product (conv width 129 over 256 -> 128 outputs per row)
__global__ void slide_k(const float* __restrict__ in, size_t base, size_t sb, size_t st,
                        int Nb, int Nt, const float* __restrict__ w,
                        const float* __restrict__ bsc, float* __restrict__ out) {
  int idx = blockIdx.x * blockDim.x + threadIdx.x;
  int total = Nb * Nt * 128;
  if (idx >= total) return;
  int k = idx & 127;
  int r = idx >> 7;
  int t = r % Nt, b = r / Nt;
  const float* p = in + base + (size_t)b * sb + (size_t)t * st + k;
  float s = bsc[0];
  for (int m = 0; m < 129; ++m) s += p[m] * w[m];
  out[idx] = s;
}

__global__ void aw_k(const float* __restrict__ h2, float* __restrict__ aw) {
  int idx = blockIdx.x * blockDim.x + threadIdx.x;  // (B,127,16)
  if (idx >= 128 * 127 * 16) return;
  int v = idx & 15;
  int r = idx >> 4;
  int t = r % 127, b = r / 127;
  int ti = (t == 0) ? 0 : (t - 1);
  const float* p = h2 + ((size_t)b * NB + v * TT + ti) * DDIM;
  const float* q = h2 + ((size_t)b * NB + v * TT + (TT - 1)) * DDIM;
  float s = 0.f;
  for (int d = 0; d < DDIM; ++d) s += p[d] * q[d];
  aw[idx] = s;
}

__global__ void soft_v_k(const float* __restrict__ aw, float* __restrict__ soft) {
  int idx = blockIdx.x * blockDim.x + threadIdx.x;  // (B,127)
  if (idx >= 128 * 127) return;
  const float* p = aw + (size_t)idx * 16;
  float mx = -1e30f;
  for (int v = 0; v < VV; ++v) mx = fmaxf(mx, p[v]);
  float e[16], s = 0.f;
  for (int v = 0; v < VV; ++v) { e[v] = __expf(p[v] - mx); s += e[v]; }
  for (int v = 0; v < VV; ++v) soft[(size_t)idx * 16 + v] = e[v] / s;
}

__global__ void energy_k(const float* __restrict__ h_t, const float* __restrict__ h_i,
                         const float* __restrict__ c1, const float* __restrict__ c2,
                         const float* __restrict__ soft,
                         const float* __restrict__ wa, const float* __restrict__ wab,
                         const float* __restrict__ wb, const float* __restrict__ wbb,
                         const float* __restrict__ wc, const float* __restrict__ wcb,
                         const float* __restrict__ wd, const float* __restrict__ wdb,
                         float* __restrict__ energy) {   // (B,16,127)
  int idx = blockIdx.x * blockDim.x + threadIdx.x;
  if (idx >= 128 * 127 * 16) return;
  int v = idx & 15;
  int r = idx >> 4;
  int t = r % 127, b = r / 127;
  float s = wab[v] + wbb[v] + wcb[v] + wdb[v];
  const float* ht = h_t + (size_t)b * 128;
  const float* hi = h_i + ((size_t)b * 127 + t) * 128;
  const float* p1 = c1 + (size_t)b * 128;
  const float* p2 = c2 + ((size_t)b * 127 + t) * 128;
  for (int k = 0; k < 128; ++k)
    s += ht[k] * wa[v * 128 + k] + hi[k] * wb[v * 128 + k] + (p1[k] + p2[k]) * wc[v * 128 + k];
  const float* sf = soft + ((size_t)b * 127 + t) * 16;
  for (int u = 0; u < VV; ++u) s += sf[u] * wd[v * 16 + u];
  energy[((size_t)b * 16 + v) * 127 + t] = tanhf(s);
}

__global__ __launch_bounds__(128)
void focus_k(const float* __restrict__ energy, float* __restrict__ attn) {
  int b = blockIdx.x, t = threadIdx.x;
  __shared__ float red[128];
  float ev[16];
  float bt = 0.f, ss = 0.f;
  if (t < 127) {
    for (int v = 0; v < VV; ++v) {
      float e = energy[((size_t)b * 16 + v) * 127 + t];
      ev[v] = e;
      bt += fmaxf(e, 0.f);
      ss += sigf(e);
    }
  }
  red[t] = (t < 127) ? bt : 0.f;
  __syncthreads();
  for (int o = 64; o > 0; o >>= 1) { if (t < o) red[t] += red[t + o]; __syncthreads(); }
  float btot = red[0] + 1e-5f;
  __syncthreads();
  float beta = bt / btot;
  float eh[16], mx = -1e30f;
  if (t < 127) {
    for (int v = 0; v < VV; ++v) {
      eh[v] = sigf(ev[v]) / (ss + 1e-5f) * beta * 2.0f;
      mx = fmaxf(mx, eh[v]);
    }
  }
  red[t] = (t < 127) ? mx : -1e30f;
  __syncthreads();
  for (int o = 64; o > 0; o >>= 1) { if (t < o) red[t] = fmaxf(red[t], red[t + o]); __syncthreads(); }
  float gmx = red[0];
  __syncthreads();
  float le = 0.f;
  if (t < 127) {
    for (int v = 0; v < VV; ++v) { eh[v] = __expf(eh[v] - gmx); le += eh[v]; }
  }
  red[t] = (t < 127) ? le : 0.f;
  __syncthreads();
  for (int o = 64; o > 0; o >>= 1) { if (t < o) red[t] += red[t + o]; __syncthreads(); }
  float tot = red[0];
  if (t < 127)
    for (int v = 0; v < VV; ++v)
      attn[((size_t)b * 16 + v) * 127 + t] = eh[v] / tot;
}

__global__ void ctx_k(const float* __restrict__ h2, const float* __restrict__ attn,
                      float* __restrict__ ctx) {   // (B,16,256)
  int idx = blockIdx.x * blockDim.x + threadIdx.x;
  if (idx >= 128 * 16 * 256) return;
  int d = idx & 255;
  int r = idx >> 8;
  int v = r & 15, b = r >> 4;
  const float* a = attn + ((size_t)b * 16 + v) * 127;
  const float* p = h2 + ((size_t)b * NB + v * TT) * DDIM + d;
  float s = 0.f;
  for (int t = 0; t < 127; ++t) s += a[t] * p[(size_t)t * DDIM];
  ctx[idx] = s;
}

__global__ void cat_k(const float* __restrict__ h2, const float* __restrict__ ctx,
                      float* __restrict__ cat) {   // (B,2,16,256)
  int idx = blockIdx.x * blockDim.x + threadIdx.x;
  if (idx >= 128 * 2 * 16 * 256) return;
  int d = idx & 255;
  int r = idx >> 8;
  int v = r & 15; r >>= 4;
  int ch = r & 1; int b = r >> 1;
  cat[idx] = ch ? ctx[((size_t)b * 16 + v) * 256 + d]
               : h2[((size_t)b * NB + v * TT + (TT - 1)) * DDIM + d];
}

// fused conv5x5(pad2) + relu + maxpool2x2
__global__ void conv1_k(const float* __restrict__ in, const float* __restrict__ w,
                        const float* __restrict__ bias, float* __restrict__ out) {
  // in (B,2,16,256) -> out (B,16,8,128)
  int idx = blockIdx.x * blockDim.x + threadIdx.x;
  if (idx >= 128 * 16 * 8 * 128) return;
  int x = idx & 127;
  int r = idx >> 7;
  int y = r & 7; r >>= 3;
  int oc = r & 15; int b = r >> 4;
  float best = -1e30f;
  for (int dy = 0; dy < 2; ++dy)
    for (int dx = 0; dx < 2; ++dx) {
      int oy = 2 * y + dy, ox = 2 * x + dx;
      float s = bias[oc];
      for (int ic = 0; ic < 2; ++ic)
        for (int ky = 0; ky < 5; ++ky) {
          int iy = oy + ky - 2;
          if (iy < 0 || iy >= 16) continue;
          for (int kx = 0; kx < 5; ++kx) {
            int ix = ox + kx - 2;
            if (ix < 0 || ix >= 256) continue;
            s += in[(((size_t)b * 2 + ic) * 16 + iy) * 256 + ix] * w[((oc * 2 + ic) * 5 + ky) * 5 + kx];
          }
        }
      best = fmaxf(best, s);
    }
  out[idx] = fmaxf(best, 0.f);
}

__global__ void conv2_k(const float* __restrict__ in, const float* __restrict__ w,
                        const float* __restrict__ bias, float* __restrict__ out) {
  // in (B,16,8,128) -> out (B,32,4,64)
  int idx = blockIdx.x * blockDim.x + threadIdx.x;
  if (idx >= 128 * 32 * 4 * 64) return;
  int x = idx & 63;
  int r = idx >> 6;
  int y = r & 3; r >>= 2;
  int oc = r & 31; int b = r >> 5;
  float best = -1e30f;
  for (int dy = 0; dy < 2; ++dy)
    for (int dx = 0; dx < 2; ++dx) {
      int oy = 2 * y + dy, ox = 2 * x + dx;
      float s = bias[oc];
      for (int ic = 0; ic < 16; ++ic)
        for (int ky = 0; ky < 5; ++ky) {
          int iy = oy + ky - 2;
          if (iy < 0 || iy >= 8) continue;
          for (int kx = 0; kx < 5; ++kx) {
            int ix = ox + kx - 2;
            if (ix < 0 || ix >= 128) continue;
            s += in[(((size_t)b * 16 + ic) * 8 + iy) * 128 + ix] * w[((oc * 16 + ic) * 5 + ky) * 5 + kx];
          }
        }
      best = fmaxf(best, s);
    }
  out[idx] = fmaxf(best, 0.f);
}

__global__ void affine_k(const float* __restrict__ z, const float* __restrict__ w,
                         const float* __restrict__ bias, float* __restrict__ out) {
  int idx = blockIdx.x * blockDim.x + threadIdx.x;  // B*C = 1280
  if (idx >= 128 * 10) return;
  int c = idx % 10, b = idx / 10;
  const float* zp = z + (size_t)b * 8192;
  const float* wp = w + (size_t)c * 8192;
  float s = bias[c];
  for (int i = 0; i < 8192; ++i) s += zp[i] * wp[i];
  out[idx] = s;
}

// ---------------------------------------------------------------------------
// Host launcher
// ---------------------------------------------------------------------------
extern "C" void kernel_launch(void* const* d_in, const int* in_sizes, int n_in,
                              void* d_out, int out_size, void* d_ws, size_t ws_size,
                              hipStream_t stream) {
  (void)in_sizes; (void)n_in; (void)out_size; (void)ws_size;
  const float* x     = (const float*)d_in[0];
  const float* wih0f = (const float*)d_in[1];
  const float* whh0f = (const float*)d_in[2];
  const float* bih0f = (const float*)d_in[3];
  const float* bhh0f = (const float*)d_in[4];
  const float* wih0b = (const float*)d_in[5];
  const float* whh0b = (const float*)d_in[6];
  const float* bih0b = (const float*)d_in[7];
  const float* bhh0b = (const float*)d_in[8];
  const float* wih1f = (const float*)d_in[9];
  const float* whh1f = (const float*)d_in[10];
  const float* bih1f = (const float*)d_in[11];
  const float* bhh1f = (const float*)d_in[12];
  const float* wih1b = (const float*)d_in[13];
  const float* whh1b = (const float*)d_in[14];
  const float* bih1b = (const float*)d_in[15];
  const float* bhh1b = (const float*)d_in[16];
  const float* wsct_w = (const float*)d_in[17];
  const float* wsct_b = (const float*)d_in[18];
  const float* wsci_w = (const float*)d_in[19];
  const float* wsci_b = (const float*)d_in[20];
  const float* wcc1_w = (const float*)d_in[21];
  const float* wcc1_b = (const float*)d_in[22];
  const float* wcc2_w = (const float*)d_in[23];
  const float* wcc2_b = (const float*)d_in[24];
  const float* wa  = (const float*)d_in[25];
  const float* wab = (const float*)d_in[26];
  const float* wb  = (const float*)d_in[27];
  const float* wbb = (const float*)d_in[28];
  const float* wc  = (const float*)d_in[29];
  const float* wcb = (const float*)d_in[30];
  const float* wd  = (const float*)d_in[31];
  const float* wdb = (const float*)d_in[32];
  const float* f1w = (const float*)d_in[33];
  const float* f1b = (const float*)d_in[34];
  const float* f2w = (const float*)d_in[35];
  const float* f2b = (const float*)d_in[36];
  const float* affw = (const float*)d_in[37];
  const float* affb = (const float*)d_in[38];

  char* ws = (char*)d_ws;
  size_t off = 0;
  auto alloc = [&](size_t bytes) -> char* {
    char* p = ws + off;
    off += (bytes + 255) & ~(size_t)255;
    return p;
  };
  unsigned short* h1bf = (unsigned short*)alloc((size_t)S_LEN * NB * DDIM * 2);
  float* h2            = (float*)alloc((size_t)S_LEN * NB * DDIM * 4);
  unsigned short* wih2f_bf = (unsigned short*)alloc((size_t)512 * 256 * 2);
  unsigned short* wih2b_bf = (unsigned short*)alloc((size_t)512 * 256 * 2);
  float* hsum_last = (float*)alloc((size_t)128 * 256 * 4);
  float* hsum_i    = (float*)alloc((size_t)128 * 127 * 256 * 4);
  float* h_t  = (float*)alloc((size_t)128 * 128 * 4);
  float* h_i  = (float*)alloc((size_t)128 * 127 * 128 * 4);
  float* hcc1 = (float*)alloc((size_t)128 * 128 * 4);
  float* hcc2 = (float*)alloc((size_t)128 * 127 * 128 * 4);
  float* awb  = (float*)alloc((size_t)128 * 127 * 16 * 4);
  float* soft = (float*)alloc((size_t)128 * 127 * 16 * 4);
  float* energy = (float*)alloc((size_t)128 * 16 * 127 * 4);
  float* attn   = (float*)alloc((size_t)128 * 16 * 127 * 4);
  float* ctxb   = (float*)alloc((size_t)128 * 16 * 256 * 4);
  float* catb   = (float*)alloc((size_t)128 * 2 * 16 * 256 * 4);
  float* z1p    = (float*)alloc((size_t)128 * 16 * 8 * 128 * 4);
  float* z2p    = (float*)alloc((size_t)128 * 32 * 4 * 64 * 4);

  // pre-convert layer-2 input weights to bf16 (n-major)
  f32_to_bf16_k<<<(131072 + 255) / 256, 256, 0, stream>>>(wih1f, wih2f_bf, 131072);
  f32_to_bf16_k<<<(131072 + 255) / 256, 256, 0, stream>>>(wih1b, wih2b_bf, 131072);

  // LSTM layers (both directions in one grid via blockIdx.y)
  lstm_l1_k<<<dim3(128, 2), 256, 0, stream>>>(x, whh0f, whh0b, wih0f, wih0b,
                                              bih0f, bhh0f, bih0b, bhh0b, h1bf);
  lstm_l2_k<<<dim3(128, 2), 256, 0, stream>>>(h1bf, whh1f, whh1b, wih2f_bf, wih2b_bf,
                                              bih1f, bhh1f, bih1b, bhh1b, h2);

  // attention tail
  sum_last_k<<<128, 256, 0, stream>>>(h2, hsum_last);
  sum_i_k<<<16256, 256, 0, stream>>>(h2, hsum_i);
  slide_k<<<64, 256, 0, stream>>>(hsum_last, 0, 256, 0, 128, 1, wsct_w, wsct_b, h_t);
  slide_k<<<8128, 256, 0, stream>>>(hsum_i, 0, (size_t)127 * 256, 256, 128, 127, wsci_w, wsci_b, h_i);
  slide_k<<<64, 256, 0, stream>>>(h2, (size_t)(15 * 128 + 127) * 256, (size_t)NB * 256, 0,
                                  128, 1, wcc1_w, wcc1_b, hcc1);
  slide_k<<<8128, 256, 0, stream>>>(h2, (size_t)(15 * 128) * 256, (size_t)NB * 256, 256,
                                    128, 127, wcc2_w, wcc2_b, hcc2);
  aw_k<<<1016, 256, 0, stream>>>(h2, awb);
  soft_v_k<<<(128 * 127 + 255) / 256, 256, 0, stream>>>(awb, soft);
  energy_k<<<1016, 256, 0, stream>>>(h_t, h_i, hcc1, hcc2, soft,
                                     wa, wab, wb, wbb, wc, wcb, wd, wdb, energy);
  focus_k<<<128, 128, 0, stream>>>(energy, attn);
  ctx_k<<<2048, 256, 0, stream>>>(h2, attn, ctxb);
  cat_k<<<4096, 256, 0, stream>>>(h2, ctxb, catb);
  conv1_k<<<8192, 256, 0, stream>>>(catb, f1w, f1b, z1p);
  conv2_k<<<4096, 256, 0, stream>>>(z1p, f2w, f2b, z2p);
  affine_k<<<5, 256, 0, stream>>>(z2p, affw, affb, (float*)d_out);
}
